// BinaryDiff_54400055771143
// MI455X (gfx1250) — matrix operand order repaired and checked by reading, not compiled, not run on hardware
//
#include <hip/hip_runtime.h>

// ---------------------------------------------------------------------------
// BinaryDiff fused GEMM for gfx1250 (MI455X):
//   out[M,N] = x[M,K] @ (base[K,N] + coeff * mask[K,N])
// Fast path (needs 192MiB workspace):
//   prep_x : x -> bf16 hi/lo            [M][K]      (one-time conversion)
//   prep_w : base + c*mask -> bf16 hi/lo, transposed [N][K]
//   gemm   : double-buffered async-to-LDS staging pipelined against 3-term
//            split-bf16 WMMA (~f32 accuracy on the bf16 matrix pipe)
// Fallback path (small ws): single-pass fused kernel (converts in-loop).
// ---------------------------------------------------------------------------

typedef __attribute__((ext_vector_type(16))) __bf16         bf16x16;
typedef __attribute__((ext_vector_type(8)))  float          f32x8;
typedef __attribute__((ext_vector_type(4)))  float          f32x4;
typedef __attribute__((ext_vector_type(4)))  int            i32x4;
typedef __attribute__((ext_vector_type(4)))  unsigned short u16x4;
typedef __attribute__((ext_vector_type(8)))  unsigned short u16x8;
typedef __attribute__((ext_vector_type(16))) unsigned short u16x16;

// address-space qualified vector types for the async-to-LDS builtin
typedef __attribute__((address_space(1))) i32x4 as1_i32x4;  // global
typedef __attribute__((address_space(3))) i32x4 as3_i32x4;  // LDS

#define M_TOT 8192
#define N_TOT 4096
#define K_TOT 4096

#define BM 128
#define BN 128
#define BK 64
#define LDR (BK + 8)   // LDS row length (u16); 144B rows keep 16B align + skew banks

#if defined(__has_builtin)
#if __has_builtin(__builtin_amdgcn_global_load_async_to_lds_b128)
#define HAVE_ASYNC_LDS 1
#endif
#endif

// round-to-nearest-even f32 -> bf16 (raw u16)
__device__ __forceinline__ unsigned short f2bf(float f) {
  unsigned int u = __float_as_uint(f);
  u += 0x7FFFu + ((u >> 16) & 1u);
  return (unsigned short)(u >> 16);
}
__device__ __forceinline__ float bf2f(unsigned short s) {
  return __uint_as_float(((unsigned int)s) << 16);
}

// build a 16-elem bf16 fragment from two 8-elem (16B) LDS chunks
__device__ __forceinline__ bf16x16 frag2(const unsigned short* p0,
                                         const unsigned short* p1) {
  u16x8 a = *(const u16x8*)p0;
  u16x8 b = *(const u16x8*)p1;
  u16x16 v = __builtin_shufflevector(a, b, 0, 1, 2, 3, 4, 5, 6, 7,
                                           8, 9, 10, 11, 12, 13, 14, 15);
  return __builtin_bit_cast(bf16x16, v);
}

// 16B global -> LDS copy; async data-mover path when the builtin exists
__device__ __forceinline__ void copy16(unsigned short* lds, const unsigned short* g) {
#if defined(HAVE_ASYNC_LDS)
  __builtin_amdgcn_global_load_async_to_lds_b128(
      (as1_i32x4*)g, (as3_i32x4*)lds, 0, 0);
#else
  *(u16x8*)lds = *(const u16x8*)g;
#endif
}

__device__ __forceinline__ void wait_stage() {
#if defined(HAVE_ASYNC_LDS)
#if __has_builtin(__builtin_amdgcn_s_wait_asynccnt)
  __builtin_amdgcn_s_wait_asynccnt(0);
#else
  asm volatile("s_wait_asynccnt 0" ::: "memory");
#endif
#endif
}

// ---------------------------------------------------------------------------
// prep_x: decompose x (f32) into bf16 hi/lo, one f32x4 per thread
// ---------------------------------------------------------------------------
__global__ __launch_bounds__(256)
void prep_x(const float* __restrict__ X,
            unsigned short* __restrict__ Xhi,
            unsigned short* __restrict__ Xlo) {
  size_t i = ((size_t)blockIdx.x * 256 + threadIdx.x) * 4;
  f32x4 v = *(const f32x4*)(X + i);
  u16x4 h, l;
#pragma unroll
  for (int j = 0; j < 4; ++j) {
    unsigned short hh = f2bf(v[j]);
    h[j] = hh;
    l[j] = f2bf(v[j] - bf2f(hh));
  }
  *(u16x4*)(Xhi + i) = h;
  *(u16x4*)(Xlo + i) = l;
}

// ---------------------------------------------------------------------------
// prep_w: W = base + c*mask, decomposed to bf16 hi/lo and TRANSPOSED to [N][K]
// (64x64 tile through LDS so global writes stay coalesced)
// ---------------------------------------------------------------------------
__global__ __launch_bounds__(256)
void prep_w(const float* __restrict__ Bw, const int* __restrict__ Mk,
            const float* __restrict__ Cf,
            unsigned short* __restrict__ Wthi,
            unsigned short* __restrict__ Wtlo) {
  __shared__ unsigned short tHi[64][72];
  __shared__ unsigned short tLo[64][72];
  const float c = Cf[0];
  const int tid = threadIdx.x;
  const int k0 = blockIdx.x * 64;
  const int n0 = blockIdx.y * 64;

#pragma unroll
  for (int it = 0; it < 4; ++it) {
    int idx = it * 256 + tid;
    int kr = idx >> 4;          // 0..63  K row
    int nc = (idx & 15) << 2;   // 0..60  N col (x4)
    f32x4 bv = *(const f32x4*)(Bw + (size_t)(k0 + kr) * N_TOT + n0 + nc);
    i32x4 mv = *(const i32x4*)(Mk + (size_t)(k0 + kr) * N_TOT + n0 + nc);
#pragma unroll
    for (int j = 0; j < 4; ++j) {
      float w = bv[j] + c * (float)mv[j];
      unsigned short h = f2bf(w);
      tHi[nc + j][kr] = h;      // transpose on LDS store
      tLo[nc + j][kr] = f2bf(w - bf2f(h));
    }
  }
  __syncthreads();
#pragma unroll
  for (int it = 0; it < 2; ++it) {
    int idx = it * 256 + tid;
    int nr = idx >> 3;          // 0..63  N row
    int kc = (idx & 7) << 3;    // 0..56  K col (x8)
    *(u16x8*)(Wthi + (size_t)(n0 + nr) * K_TOT + k0 + kc) = *(const u16x8*)&tHi[nr][kc];
    *(u16x8*)(Wtlo + (size_t)(n0 + nr) * K_TOT + k0 + kc) = *(const u16x8*)&tLo[nr][kc];
  }
}

// ---------------------------------------------------------------------------
// Fast GEMM: pre-decomposed bf16 operands; double-buffered async staging
// pipelined against the WMMA compute.
// ---------------------------------------------------------------------------
__global__ __launch_bounds__(256)
void bdiff_wmma_pre(const unsigned short* __restrict__ Xhi,
                    const unsigned short* __restrict__ Xlo,
                    const unsigned short* __restrict__ Wthi,  // [N][K]
                    const unsigned short* __restrict__ Wtlo,  // [N][K]
                    float* __restrict__ Out) {
  __shared__ unsigned short sXhi[2][BM][LDR];
  __shared__ unsigned short sXlo[2][BM][LDR];
  __shared__ unsigned short sWhi[2][BN][LDR];
  __shared__ unsigned short sWlo[2][BN][LDR];

  const int tid  = threadIdx.x;
  const int lane = tid & 31;
  const int wave = tid >> 5;

  const int m0 = blockIdx.x * BM;   // M fastest: W column-group stays L2-hot
  const int n0 = blockIdx.y * BN;
  const int wm = (wave & 3) * 32;   // 4 (M) x 2 (N) wave grid, 32x64 per wave
  const int wn = (wave >> 2) * 64;

  f32x8 acc[2][4] = {};

  // stage one k-tile into LDS buffer `buf`
  auto stage = [&](int buf, int k0) {
#pragma unroll
    for (int it = 0; it < 4; ++it) {
      int idx = it * 256 + tid;
      int r  = idx >> 3;          // 0..127
      int kc = (idx & 7) << 3;    // 0..56 step 8
      copy16(&sXhi[buf][r][kc], Xhi + (size_t)(m0 + r) * K_TOT + k0 + kc);
      copy16(&sXlo[buf][r][kc], Xlo + (size_t)(m0 + r) * K_TOT + k0 + kc);
      copy16(&sWhi[buf][r][kc], Wthi + (size_t)(n0 + r) * K_TOT + k0 + kc);
      copy16(&sWlo[buf][r][kc], Wtlo + (size_t)(n0 + r) * K_TOT + k0 + kc);
    }
  };

  // prologue: fill buffer 0
  stage(0, 0);
  wait_stage();
  __syncthreads();

  int buf = 0;
  for (int k0 = 0; k0 < K_TOT; k0 += BK) {
    // kick off the next tile into the other buffer (async engine overlaps WMMA)
    if (k0 + BK < K_TOT) stage(buf ^ 1, k0 + BK);

    const unsigned short (*pXhi)[LDR] = sXhi[buf];
    const unsigned short (*pXlo)[LDR] = sXlo[buf];
    const unsigned short (*pWhi)[LDR] = sWhi[buf];
    const unsigned short (*pWlo)[LDR] = sWlo[buf];

#pragma unroll
    for (int ks = 0; ks < BK; ks += 32) {
      const int ml = lane & 15;
      const int ka = (lane >> 4) << 3;   // A K half-offset 0 / 8
      const int kb = (lane >> 4) << 3;   // B K half-offset 0 / 8

      bf16x16 ahi[2], alo[2], bhi[4], blo[4];
#pragma unroll
      for (int mi = 0; mi < 2; ++mi) {
        const unsigned short* rh = &pXhi[wm + mi * 16 + ml][ks];
        const unsigned short* rl = &pXlo[wm + mi * 16 + ml][ks];
        ahi[mi] = frag2(rh + ka, rh + 16 + ka);
        alo[mi] = frag2(rl + ka, rl + 16 + ka);
      }
#pragma unroll
      for (int ni = 0; ni < 4; ++ni) {
        const unsigned short* rh = &pWhi[wn + ni * 16 + ml][ks];
        const unsigned short* rl = &pWlo[wn + ni * 16 + ml][ks];
        bhi[ni] = frag2(rh + kb, rh + 16 + kb);
        blo[ni] = frag2(rl + kb, rl + 16 + kb);
      }
#pragma unroll
      for (int mi = 0; mi < 2; ++mi) {
#pragma unroll
        for (int ni = 0; ni < 4; ++ni) {
          acc[mi][ni] = __builtin_amdgcn_wmma_f32_16x16x32_bf16(
              false, ahi[mi], false, bhi[ni], (short)0, acc[mi][ni], false, false);
          acc[mi][ni] = __builtin_amdgcn_wmma_f32_16x16x32_bf16(
              false, ahi[mi], false, blo[ni], (short)0, acc[mi][ni], false, false);
          acc[mi][ni] = __builtin_amdgcn_wmma_f32_16x16x32_bf16(
              false, alo[mi], false, bhi[ni], (short)0, acc[mi][ni], false, false);
        }
      }
    }

    // drain the async stage for the next tile, then flip buffers
    wait_stage();
    __syncthreads();
    buf ^= 1;
  }

  // ---- epilogue: VGPR r -> M = r (lanes 0-15) / r+8 (lanes 16-31) ----
  {
    const int ml = lane & 15;
    const int rb = (lane >> 4) << 3;
#pragma unroll
    for (int mi = 0; mi < 2; ++mi) {
#pragma unroll
      for (int ni = 0; ni < 4; ++ni) {
        const int col = n0 + wn + ni * 16 + ml;
        const int row = m0 + wm + mi * 16 + rb;
#pragma unroll
        for (int r = 0; r < 8; ++r) {
          Out[(size_t)(row + r) * N_TOT + col] = acc[mi][ni][r];
        }
      }
    }
  }
}

// ---------------------------------------------------------------------------
// Fallback: single-pass fused kernel (converts inside the k-loop)
// ---------------------------------------------------------------------------
__global__ __launch_bounds__(256)
void binary_diff_gemm(const float* __restrict__ X, const float* __restrict__ Bw,
                      const float* __restrict__ Cf, const int* __restrict__ Mk,
                      float* __restrict__ Out) {
  __shared__ unsigned short sXhi[BM][LDR];
  __shared__ unsigned short sXlo[BM][LDR];
  __shared__ unsigned short sWhi[BN][LDR];
  __shared__ unsigned short sWlo[BN][LDR];

  const int tid  = threadIdx.x;
  const int lane = tid & 31;
  const int wave = tid >> 5;
  const float c  = Cf[0];

  const int m0 = blockIdx.x * BM;
  const int n0 = blockIdx.y * BN;
  const int wm = (wave & 3) * 32;
  const int wn = (wave >> 2) * 64;

  f32x8 acc[2][4] = {};

  for (int k0 = 0; k0 < K_TOT; k0 += BK) {
    __syncthreads();
#pragma unroll
    for (int it = 0; it < 8; ++it) {
      int idx = it * 256 + tid;
      int r   = idx >> 4;
      int cc  = (idx & 15) << 2;
      f32x4 xv = *(const f32x4*)(X + (size_t)(m0 + r) * K_TOT + k0 + cc);
      u16x4 hv, lv;
#pragma unroll
      for (int j = 0; j < 4; ++j) {
        unsigned short h = f2bf(xv[j]);
        hv[j] = h;
        lv[j] = f2bf(xv[j] - bf2f(h));
      }
      *(u16x4*)&sXhi[r][cc] = hv;
      *(u16x4*)&sXlo[r][cc] = lv;
    }
#pragma unroll
    for (int it = 0; it < 8; ++it) {
      int idx = it * 256 + tid;
      int kr  = idx >> 5;
      int nc  = (idx & 31) << 2;
      f32x4 bv = *(const f32x4*)(Bw + (size_t)(k0 + kr) * N_TOT + n0 + nc);
      i32x4 mv = *(const i32x4*)(Mk + (size_t)(k0 + kr) * N_TOT + n0 + nc);
#pragma unroll
      for (int j = 0; j < 4; ++j) {
        float w = bv[j] + c * (float)mv[j];
        unsigned short h = f2bf(w);
        sWhi[nc + j][kr] = h;
        sWlo[nc + j][kr] = f2bf(w - bf2f(h));
      }
    }
    __syncthreads();

#pragma unroll
    for (int ks = 0; ks < BK; ks += 32) {
      const int ml = lane & 15;
      const int ka = (lane >> 4) << 3;
      const int kb = (lane >> 4) << 3;
      bf16x16 ahi[2], alo[2], bhi[4], blo[4];
#pragma unroll
      for (int mi = 0; mi < 2; ++mi) {
        const unsigned short* rh = &sXhi[wm + mi * 16 + ml][ks];
        const unsigned short* rl = &sXlo[wm + mi * 16 + ml][ks];
        ahi[mi] = frag2(rh + ka, rh + 16 + ka);
        alo[mi] = frag2(rl + ka, rl + 16 + ka);
      }
#pragma unroll
      for (int ni = 0; ni < 4; ++ni) {
        const unsigned short* rh = &sWhi[wn + ni * 16 + ml][ks];
        const unsigned short* rl = &sWlo[wn + ni * 16 + ml][ks];
        bhi[ni] = frag2(rh + kb, rh + 16 + kb);
        blo[ni] = frag2(rl + kb, rl + 16 + kb);
      }
#pragma unroll
      for (int mi = 0; mi < 2; ++mi) {
#pragma unroll
        for (int ni = 0; ni < 4; ++ni) {
          acc[mi][ni] = __builtin_amdgcn_wmma_f32_16x16x32_bf16(
              false, ahi[mi], false, bhi[ni], (short)0, acc[mi][ni], false, false);
          acc[mi][ni] = __builtin_amdgcn_wmma_f32_16x16x32_bf16(
              false, ahi[mi], false, blo[ni], (short)0, acc[mi][ni], false, false);
          acc[mi][ni] = __builtin_amdgcn_wmma_f32_16x16x32_bf16(
              false, alo[mi], false, bhi[ni], (short)0, acc[mi][ni], false, false);
        }
      }
    }
  }

  {
    const int ml = lane & 15;
    const int rb = (lane >> 4) << 3;
#pragma unroll
    for (int mi = 0; mi < 2; ++mi) {
#pragma unroll
      for (int ni = 0; ni < 4; ++ni) {
        const int col = n0 + wn + ni * 16 + ml;
        const int row = m0 + wm + mi * 16 + rb;
#pragma unroll
        for (int r = 0; r < 8; ++r) {
          Out[(size_t)(row + r) * N_TOT + col] = acc[mi][ni][r];
        }
      }
    }
  }
}

extern "C" void kernel_launch(void* const* d_in, const int* in_sizes, int n_in,
                              void* d_out, int out_size, void* d_ws, size_t ws_size,
                              hipStream_t stream) {
  (void)in_sizes; (void)n_in; (void)out_size;
  const float* x     = (const float*)d_in[0];
  const float* base  = (const float*)d_in[1];
  const float* coeff = (const float*)d_in[2];
  const int*   mask  = (const int*)d_in[3];
  float*       out   = (float*)d_out;

  const size_t MK = (size_t)M_TOT * K_TOT;
  const size_t NK = (size_t)N_TOT * K_TOT;
  const size_t need = (2 * MK + 2 * NK) * sizeof(unsigned short);  // 192 MiB

  if (ws_size >= need) {
    unsigned short* Xhi  = (unsigned short*)d_ws;
    unsigned short* Xlo  = Xhi + MK;
    unsigned short* Wthi = Xlo + MK;
    unsigned short* Wtlo = Wthi + NK;

    hipLaunchKernelGGL(prep_x, dim3((unsigned)(MK / (4 * 256))), dim3(256), 0, stream,
                       x, Xhi, Xlo);
    hipLaunchKernelGGL(prep_w, dim3(K_TOT / 64, N_TOT / 64), dim3(256), 0, stream,
                       base, mask, coeff, Wthi, Wtlo);
    hipLaunchKernelGGL(bdiff_wmma_pre, dim3(M_TOT / BM, N_TOT / BN), dim3(256), 0, stream,
                       Xhi, Xlo, Wthi, Wtlo, out);
  } else {
    hipLaunchKernelGGL(binary_diff_gemm, dim3(M_TOT / BM, N_TOT / BN), dim3(256), 0, stream,
                       x, base, coeff, mask, out);
  }
}
